// EGConv_layer_72688026518113
// MI455X (gfx1250) — compile-verified, matrix operand order checked by reading, not run on hardware
//
#include <hip/hip_runtime.h>

// EGConv (symnorm/sum/max aggregators, B=4 bases, H=8 heads) + GraphNorm + ReLU
// N=50000, E=800000, D=256, F=32, B*F=128, H*B*A=96, G=64. fp32 in/out.
// Bandwidth-bound on the edge gather/scatter; GEMMs on v_wmma_f32_16x16x32_f16.

#define D_DIM 256
#define BF    128   // B*F  (bases width)
#define HBA    96   // H*B*A (comb width)
#define G_DIM  64   // graphs in batch (reference constant)

typedef __attribute__((ext_vector_type(16))) _Float16 v16h;
typedef __attribute__((ext_vector_type(8)))  float    v8f;

// order-preserving float<->uint map so unsigned atomicMax == float max
__device__ __forceinline__ unsigned mapf(float x) {
  unsigned b = __float_as_uint(x);
  return (b & 0x80000000u) ? ~b : (b | 0x80000000u);
}
__device__ __forceinline__ float unmapf(unsigned u) {
  return __uint_as_float((u & 0x80000000u) ? (u ^ 0x80000000u) : ~u);
}

__global__ void fill_f32(float* __restrict__ p, float v, size_t n) {
  size_t i = (size_t)blockIdx.x * blockDim.x + threadIdx.x;
  if (i < n) p[i] = v;
}

__global__ void deg_accum(const int* __restrict__ dst, float* __restrict__ deg, int E) {
  int i = blockIdx.x * blockDim.x + threadIdx.x;
  if (i < E) atomicAdd(&deg[dst[i]], 1.0f);
}

__global__ void deg_to_dinv(float* __restrict__ deg, int n) {
  int i = blockIdx.x * blockDim.x + threadIdx.x;
  if (i < n) { float d = deg[i]; deg[i] = d > 0.0f ? rsqrtf(d) : 0.0f; }
}

// C[M,NOUT] = A[M,KT] @ W[KT,NOUT] (+bias). One wave per 16x32 output super-tile:
// two f32 accumulators share one A fragment (halves A traffic). KT and NOUT are
// compile-time -> fully unrolled 8 k-steps (16 static WMMAs) and all W/C strides
// fold into immediate load/store offsets off a single base address.
// Requires M%16==0, NOUT%32==0, KT%32==0 (holds: 50000 / 128,96 / 256).
template <int KT, int NOUT>
__global__ __launch_bounds__(128) void wmma_gemm(
    const float* __restrict__ A, const float* __restrict__ W,
    const float* __restrict__ bias, float* __restrict__ C, int M)
{
  constexpr int ntile2 = NOUT >> 5;                         // 32-wide col tiles
  int T = (M >> 4) * ntile2;
  int wid = (blockIdx.x * blockDim.x + threadIdx.x) >> 5;   // wave-uniform
  if (wid >= T) return;                                     // whole wave exits -> EXEC all-1s
  int tm = wid / ntile2, tn = wid % ntile2;
  int lane = threadIdx.x & 31;
  int half = lane >> 4, l15 = lane & 15;
  int row  = tm * 16 + l15;        // A fragment row for this lane
  int col0 = tn * 32 + l15;        // B/C columns for this lane
  const float* Arow = A + (size_t)row * KT;
  const float* Wcol = W + col0;    // all W loads: immediate offsets from here

  v8f acc0 = {}, acc1 = {};
#pragma unroll
  for (int k0 = 0; k0 < KT; k0 += 32) {
    v16h a, b0, b1;
#pragma unroll
    for (int v = 0; v < 8; ++v) {
      // 16-bit A 16x32 layout: VGPR v holds K = kb, kb+1 ; lanes>=16 shift K by 8
      int kb = (v < 4) ? (2 * v + 8 * half) : (16 + 2 * (v - 4) + 8 * half);
      int k = k0 + kb;
      float2 av = *(const float2*)(Arow + k);     // 8B aligned: k even, row stride 1KB
      a[2 * v]      = (_Float16)av.x;
      a[2 * v + 1]  = (_Float16)av.y;
      b0[2 * v]     = (_Float16)Wcol[(size_t)k * NOUT];
      b0[2 * v + 1] = (_Float16)Wcol[(size_t)(k + 1) * NOUT];
      b1[2 * v]     = (_Float16)Wcol[(size_t)k * NOUT + 16];
      b1[2 * v + 1] = (_Float16)Wcol[(size_t)(k + 1) * NOUT + 16];
    }
    acc0 = __builtin_amdgcn_wmma_f32_16x16x32_f16(
        false, a, false, b0, (short)0, acc0, false, false);
    acc1 = __builtin_amdgcn_wmma_f32_16x16x32_f16(
        false, a, false, b1, (short)0, acc1, false, false);
  }
  float bi0 = bias ? bias[col0] : 0.0f;
  float bi1 = bias ? bias[col0 + 16] : 0.0f;
  float* Crow = C + (size_t)(tm * 16 + 8 * half) * NOUT + col0;
#pragma unroll
  for (int v = 0; v < 8; ++v) {                  // C layout: VGPR v -> row v + 8*half
    Crow[(size_t)v * NOUT]      = acc0[v] + bi0;
    Crow[(size_t)v * NOUT + 16] = acc1[v] + bi1;
  }
}

// One wave per edge: gather 128-float bases row of src (float4/lane, coalesced),
// scatter into 3 aggregators at dst via atomics.
__global__ __launch_bounds__(256) void edge_scatter(
    const int* __restrict__ src, const int* __restrict__ dst,
    const float* __restrict__ dinv, const float* __restrict__ bases,
    float* __restrict__ aggS, float* __restrict__ aggU,
    unsigned* __restrict__ aggM, int E)
{
  int wid  = (blockIdx.x * blockDim.x + threadIdx.x) >> 5;
  int lane = threadIdx.x & 31;
  if (wid >= E) return;
  int s = src[wid], d = dst[wid];
  float w = dinv[s] * dinv[d];
  float4 m = ((const float4*)(bases + (size_t)s * BF))[lane];
  size_t o = (size_t)d * BF + (size_t)lane * 4;
  atomicAdd(aggS + o + 0, m.x * w); atomicAdd(aggS + o + 1, m.y * w);
  atomicAdd(aggS + o + 2, m.z * w); atomicAdd(aggS + o + 3, m.w * w);
  atomicAdd(aggU + o + 0, m.x);     atomicAdd(aggU + o + 1, m.y);
  atomicAdd(aggU + o + 2, m.z);     atomicAdd(aggU + o + 3, m.w);
  atomicMax(aggM + o + 0, mapf(m.x)); atomicMax(aggM + o + 1, mapf(m.y));
  atomicMax(aggM + o + 2, mapf(m.z)); atomicMax(aggM + o + 3, mapf(m.w));
}

// Self-loop contributions (src == dst == node, w = dinv[n]^2).
__global__ __launch_bounds__(256) void self_scatter(
    const float* __restrict__ dinv, const float* __restrict__ bases,
    float* __restrict__ aggS, float* __restrict__ aggU,
    unsigned* __restrict__ aggM, int n)
{
  int wid  = (blockIdx.x * blockDim.x + threadIdx.x) >> 5;
  int lane = threadIdx.x & 31;
  if (wid >= n) return;
  float w = dinv[wid] * dinv[wid];
  float4 m = ((const float4*)(bases + (size_t)wid * BF))[lane];
  size_t o = (size_t)wid * BF + (size_t)lane * 4;
  atomicAdd(aggS + o + 0, m.x * w); atomicAdd(aggS + o + 1, m.y * w);
  atomicAdd(aggS + o + 2, m.z * w); atomicAdd(aggS + o + 3, m.w * w);
  atomicAdd(aggU + o + 0, m.x);     atomicAdd(aggU + o + 1, m.y);
  atomicAdd(aggU + o + 2, m.z);     atomicAdd(aggU + o + 3, m.w);
  atomicMax(aggM + o + 0, mapf(m.x)); atomicMax(aggM + o + 1, mapf(m.y));
  atomicMax(aggM + o + 2, mapf(m.z)); atomicMax(aggM + o + 3, mapf(m.w));
}

// h[n, h*32+f] = sum_{k=0..11} comb[n, h*12+k] * agg_{k/4}[n, (k%4)*32+f] + bias.
// One 256-thread block per node; aggregators + comb staged in LDS.
__global__ __launch_bounds__(256) void combine(
    const float* __restrict__ comb, const float* __restrict__ aggS,
    const float* __restrict__ aggU, const unsigned* __restrict__ aggM,
    const float* __restrict__ conv_bias, float* __restrict__ h)
{
  int node = blockIdx.x;
  int t = threadIdx.x;
  __shared__ float sA[3 * BF];
  __shared__ float sW[HBA];
  if (t < BF) {
    sA[t]          = aggS[(size_t)node * BF + t];
    sA[BF + t]     = aggU[(size_t)node * BF + t];
    sA[2 * BF + t] = unmapf(aggM[(size_t)node * BF + t]);
  }
  if (t < HBA) sW[t] = comb[(size_t)node * HBA + t];
  __syncthreads();
  int head = t >> 5, f = t & 31;
  const float* wv = sW + head * 12;
  float acc = conv_bias[t];
#pragma unroll
  for (int k = 0; k < 12; ++k)        // k = a*B + b on the aggregated side
    acc += wv[k] * sA[(k >> 2) * BF + (k & 3) * 32 + f];
  h[(size_t)node * D_DIM + t] = acc;
}

__global__ void gn_count(const int* __restrict__ batch, float* __restrict__ counts, int n) {
  int i = blockIdx.x * blockDim.x + threadIdx.x;
  if (i < n) atomicAdd(&counts[batch[i]], 1.0f);
}

__global__ __launch_bounds__(256) void gn_sum(
    const float* __restrict__ h, const int* __restrict__ batch,
    float* __restrict__ msum, int n)
{
  size_t idx = (size_t)blockIdx.x * 256 + threadIdx.x;
  int node = (int)(idx >> 8), d = (int)(idx & 255);
  if (node < n) atomicAdd(&msum[batch[node] * D_DIM + d], h[idx]);
}

__global__ void gn_mean_fin(float* __restrict__ msum, const float* __restrict__ counts) {
  int i = blockIdx.x * blockDim.x + threadIdx.x;   // G*256 exact
  msum[i] /= fmaxf(counts[i >> 8], 1.0f);
}

__global__ __launch_bounds__(256) void gn_center_var(
    float* __restrict__ h, const int* __restrict__ batch,
    const float* __restrict__ mean, const float* __restrict__ alpha,
    float* __restrict__ vsum, int n)
{
  size_t idx = (size_t)blockIdx.x * 256 + threadIdx.x;
  int node = (int)(idx >> 8), d = (int)(idx & 255);
  if (node >= n) return;
  int g = batch[node];
  float hc = h[idx] - alpha[d] * mean[g * D_DIM + d];
  h[idx] = hc;
  atomicAdd(&vsum[g * D_DIM + d], hc * hc);
}

__global__ __launch_bounds__(256) void gn_normalize(
    float* __restrict__ h, const int* __restrict__ batch,
    const float* __restrict__ vsum, const float* __restrict__ counts,
    const float* __restrict__ gamma, const float* __restrict__ beta, int n)
{
  size_t idx = (size_t)blockIdx.x * 256 + threadIdx.x;
  int node = (int)(idx >> 8), d = (int)(idx & 255);
  if (node >= n) return;
  int g = batch[node];
  float var = vsum[g * D_DIM + d] / fmaxf(counts[g], 1.0f);
  float v = gamma[d] * h[idx] * rsqrtf(var + 1e-5f) + beta[d];
  h[idx] = fmaxf(v, 0.0f);
}

extern "C" void kernel_launch(void* const* d_in, const int* in_sizes, int n_in,
                              void* d_out, int out_size, void* d_ws, size_t ws_size,
                              hipStream_t stream) {
  const float* node  = (const float*)d_in[0];
  const float* Wb    = (const float*)d_in[2];   // [256,128]
  const float* Wc    = (const float*)d_in[3];   // [256,96]
  const float* bcomb = (const float*)d_in[4];   // [96]
  const float* cbias = (const float*)d_in[5];   // [256]
  const float* gnw   = (const float*)d_in[6];
  const float* gnb   = (const float*)d_in[7];
  const float* gna   = (const float*)d_in[8];
  const int*   eidx  = (const int*)d_in[9];     // [2,E]
  const int*   batch = (const int*)d_in[10];

  int N = in_sizes[0] / D_DIM;
  int E = in_sizes[9] / 2;
  const int* src = eidx;
  const int* dst = eidx + E;
  float* out = (float*)d_out;

  // workspace layout (all re-initialized each call; ~122 MB)
  float* ws = (float*)d_ws;
  size_t o = 0;
  float*    deg   = ws + o; o += (size_t)N;            // becomes dinv in place
  float*    bases = ws + o; o += (size_t)N * BF;
  float*    comb  = ws + o; o += (size_t)N * HBA;
  float*    aggS  = ws + o; o += (size_t)N * BF;       // symnorm   } contiguous,
  float*    aggU  = ws + o; o += (size_t)N * BF;       // sum       } one zero-fill
  unsigned* aggM  = (unsigned*)(ws + o); o += (size_t)N * BF; // max (mapped uint)
  float*    counts= ws + o; o += (size_t)G_DIM;        // } contiguous stats,
  float*    msum  = ws + o; o += (size_t)G_DIM * D_DIM;// } one zero-fill
  float*    vsum  = ws + o; o += (size_t)G_DIM * D_DIM;
  (void)ws_size; (void)n_in; (void)out_size; (void)vsum;

  auto cdiv = [](size_t a, size_t b) { return (unsigned)((a + b - 1) / b); };

  // init
  fill_f32<<<cdiv((size_t)N, 256), 256, 0, stream>>>(deg, 1.0f, (size_t)N); // self-loop deg
  fill_f32<<<cdiv((size_t)N * BF * 3, 256), 256, 0, stream>>>(aggS, 0.0f, (size_t)N * BF * 3);
  fill_f32<<<cdiv((size_t)G_DIM * (1 + 2 * D_DIM), 256), 256, 0, stream>>>(
      counts, 0.0f, (size_t)G_DIM * (1 + 2 * D_DIM));

  // gcn_norm degrees
  deg_accum<<<cdiv((size_t)E, 256), 256, 0, stream>>>(dst, deg, E);
  deg_to_dinv<<<cdiv((size_t)N, 256), 256, 0, stream>>>(deg, N);

  // dense GEMMs on WMMA (16x32 output per wave, K=256 fully unrolled)
  {
    size_t T1 = ((size_t)N / 16) * (BF / 32);
    wmma_gemm<D_DIM, BF><<<cdiv(T1 * 32, 128), 128, 0, stream>>>(node, Wb, nullptr, bases, N);
    size_t T2 = ((size_t)N / 16) * (HBA / 32);
    wmma_gemm<D_DIM, HBA><<<cdiv(T2 * 32, 128), 128, 0, stream>>>(node, Wc, bcomb, comb, N);
  }

  // message passing (dominant bandwidth stage)
  self_scatter<<<cdiv((size_t)N * 32, 256), 256, 0, stream>>>(deg, bases, aggS, aggU, aggM, N);
  edge_scatter<<<cdiv((size_t)E * 32, 256), 256, 0, stream>>>(src, dst, deg, bases, aggS, aggU, aggM, E);

  // per-node head combination -> h (stored in d_out)
  combine<<<N, 256, 0, stream>>>(comb, aggS, aggU, aggM, cbias, out);

  // GraphNorm + ReLU (in place on d_out)
  gn_count<<<cdiv((size_t)N, 256), 256, 0, stream>>>(batch, counts, N);
  gn_sum<<<N, 256, 0, stream>>>(out, batch, msum, N);
  gn_mean_fin<<<(G_DIM * D_DIM) / 256, 256, 0, stream>>>(msum, counts);
  gn_center_var<<<N, 256, 0, stream>>>(out, batch, msum, gna, vsum, N);
  gn_normalize<<<N, 256, 0, stream>>>(out, batch, vsum, counts, gnw, gnb, N);
}
